// MultiHeadAttention_21242908246654
// MI455X (gfx1250) — matrix operand order repaired and checked by reading, not compiled, not run on hardware
//
#include <hip/hip_runtime.h>

#define DEVFN __device__ __forceinline__

typedef __attribute__((ext_vector_type(16))) _Float16 v16h;
typedef __attribute__((ext_vector_type(8)))  float    v8f;

static constexpr int Bn   = 4;
static constexpr int Sn   = 2048;
static constexpr int En   = 512;
static constexpr int Hn   = 8;
static constexpr int DKn  = 64;
static constexpr int DVn  = 6;
static constexpr int DVP  = 16;   // padded per-head value dim (col 15 = ones)
static constexpr int CTXW = 64;   // padded concat width (H*DV = 48 -> 64)
// softmax in base 2: scores pre-scaled by sqrt(E)*log2(e) inside the Q projection
#define QSCALE (22.6274169979695f * 1.4426950408889634f)
#define EXP2F(x) __builtin_exp2f(x)   // lowers to v_exp_f32

// ---- fragment loaders -------------------------------------------------------
DEVFN v16h ldfrag_h(const _Float16* p0, const _Float16* p1) {
  union { v16h v; uint4 q[2]; } u;
  u.q[0] = *(const uint4*)p0;
  u.q[1] = *(const uint4*)p1;
  return u.v;
}
DEVFN v16h ldfrag_f(const float* p0, const float* p1) {
  v16h v;
#pragma unroll
  for (int i = 0; i < 8; ++i) {
    v[i]     = (_Float16)p0[i];
    v[i + 8] = (_Float16)p1[i];
  }
  return v;
}

DEVFN v8f wmma16(v16h a, v16h b, v8f c) {
  return __builtin_amdgcn_wmma_f32_16x16x32_f16(false, a, false, b, (short)0, c,
                                                false, false);
}

// row-max across each 16-lane row via DPP row_ror doubling (no LDS round-trips)
DEVFN float rmax16(float x) {
  int t;
  t = __builtin_amdgcn_update_dpp(__float_as_int(x), __float_as_int(x), 0x121, 0xF, 0xF, false);
  x = fmaxf(x, __int_as_float(t));
  t = __builtin_amdgcn_update_dpp(__float_as_int(x), __float_as_int(x), 0x122, 0xF, 0xF, false);
  x = fmaxf(x, __int_as_float(t));
  t = __builtin_amdgcn_update_dpp(__float_as_int(x), __float_as_int(x), 0x124, 0xF, 0xF, false);
  x = fmaxf(x, __int_as_float(t));
  t = __builtin_amdgcn_update_dpp(__float_as_int(x), __float_as_int(x), 0x128, 0xF, 0xF, false);
  x = fmaxf(x, __int_as_float(t));
  return x;
}

// ---- K0: weight transpose + f32->f16 convert with padding ------------------
// src: [H][R][C] f32 ; dst: [H][Cp][Rp] f16 ; dst[h][c][r] = (c<C && r<R) ? src[h][r][c] : 0
__global__ void wconv_kernel(const float* __restrict__ src,
                             _Float16* __restrict__ dst,
                             int H, int R, int C, int Rp, int Cp) {
  int i = blockIdx.x * blockDim.x + threadIdx.x;
  int total = H * Cp * Rp;
  if (i >= total) return;
  int r = i % Rp;
  int t = i / Rp;
  int c = t % Cp;
  int h = t / Cp;
  float v = 0.f;
  if (c < C && r < R) v = src[((size_t)h * R + r) * C + c];
  dst[i] = (_Float16)v;
}

// ---- K1: input projections --------------------------------------------------
// One wave owns ALL NT n-tiles of one (h, m-tile); fp32 A rows loaded once.
// outscale folds sqrt(E)*log2e into Qh; ones_col writes a constant-1 column (Vh).
template <int NT>
__global__ void proj_kernel(const float* __restrict__ X,      // [B*S][E]
                            const _Float16* __restrict__ WT,  // [H][Dout][E]
                            const float* __restrict__ bias,   // [H][nbias]
                            _Float16* __restrict__ out,
                            int nbias, int transposed, float outscale, int ones_col) {
  const int Dout = NT * 16;
  const int lane = threadIdx.x & 31;
  const int wave = __builtin_amdgcn_readfirstlane(threadIdx.x >> 5);  // uniform
  const int MT = (Bn * Sn) / 16;
  int gw = blockIdx.x * (blockDim.x >> 5) + wave;
  if (gw >= Hn * MT) return;
  const int h  = gw / MT;
  const int mt = gw % MT;
  const int mbase = mt * 16;
  const int b  = mbase / Sn;
  const int sb = mbase % Sn;

  const int n16   = lane & 15;
  const int aoff  = (lane < 16) ? 0 : 8;
  const int boff  = (lane < 16) ? 0 : 16;
  const int half8 = (lane < 16) ? 0 : 8;

  const float* arow = X + (size_t)(mbase + n16) * En;
  const _Float16* wbase = WT + ((size_t)h * Dout + n16) * En;

  v8f acc[NT];
#pragma unroll
  for (int nt = 0; nt < NT; ++nt) acc[nt] = (v8f){};

#pragma unroll 2
  for (int kb = 0; kb < En; kb += 32) {
    v16h a = ldfrag_f(arow + kb + aoff, arow + kb + aoff + 16);
#pragma unroll
    for (int nt = 0; nt < NT; ++nt) {
      const _Float16* brow = wbase + (size_t)nt * 16 * En + kb + half8;
      v16h bm = ldfrag_h(brow, brow + 16);
      acc[nt] = wmma16(a, bm, acc[nt]);
    }
  }

#pragma unroll
  for (int nt = 0; nt < NT; ++nt) {
    const int ncol = nt * 16 + n16;
    float bv = (ncol < nbias) ? bias[h * nbias + ncol]
                              : ((ncol == ones_col) ? 1.0f : 0.f);
    if (!transposed) {
#pragma unroll
      for (int r = 0; r < 8; ++r) {
        float v = (acc[nt][r] + bv) * outscale;
        out[(((size_t)b * Hn + h) * Sn + sb + r + half8) * Dout + ncol] = (_Float16)v;
      }
    } else {  // out: [B][H][Dout][S]
#pragma unroll
      for (int r = 0; r < 8; ++r) {
        float v = (acc[nt][r] + bv) * outscale;
        out[(((size_t)b * Hn + h) * Dout + ncol) * Sn + sb + r + half8] = (_Float16)v;
      }
    }
  }
}

// ---- K2: causal attention, single-pass flash, base-2 softmax ----------------
// Denominator rides for free in ctx column 15 (Vh ones column). Row stats use
// DPP reductions only; scores arrive pre-scaled from the Q projection. Loop
// control is fully scalar (readfirstlane wave id) so EXEC stays all-ones at
// every WMMA, per the ISA requirement.
__global__ void attn_kernel(const _Float16* __restrict__ Qh,   // [B][H][S][64] (pre-scaled)
                            const _Float16* __restrict__ Kh,   // [B][H][S][64]
                            const _Float16* __restrict__ VhT,  // [B][H][16][S]
                            _Float16* __restrict__ ctx) {      // [B*S][64]
  __shared__ __align__(16) _Float16 Pbuf[8][16][32];
  __shared__ float Ms[8][16];
  __shared__ float Cacc[16][16];

  const int lane = threadIdx.x & 31;
  const int wave = __builtin_amdgcn_readfirstlane(threadIdx.x >> 5);  // uniform
  const int tid  = threadIdx.x;

  const int qt = blockIdx.x & 127;
  const int h  = (blockIdx.x >> 7) & 7;
  const int b  = blockIdx.x >> 10;
  const int qbase = qt * 16;

  const _Float16* qb  = Qh  + (((size_t)b * Hn + h) * Sn) * DKn;
  const _Float16* kbp = Kh  + (((size_t)b * Hn + h) * Sn) * DKn;
  const _Float16* vbp = VhT + (((size_t)b * Hn + h) * DVP) * Sn;

  Cacc[tid >> 4][tid & 15] = 0.f;

  const int n16   = lane & 15;
  const int aoff  = (lane < 16) ? 0 : 8;
  const int boff  = (lane < 16) ? 0 : 16;
  const int half8 = (lane < 16) ? 0 : 8;

  const _Float16* qrow = qb + (size_t)(qbase + n16) * DKn;
  const v16h aq0 = ldfrag_h(qrow + aoff,      qrow + aoff + 16);
  const v16h aq1 = ldfrag_h(qrow + 32 + aoff, qrow + 32 + aoff + 16);

  float mreg[8];
#pragma unroll
  for (int r = 0; r < 8; ++r) mreg[r] = -1e30f;
  v8f cacc = {};

  const int pairs = (qt + 2) >> 1;   // ceil((qt+1)/2) key-tile pairs (32 keys)
  for (int p = wave; p < pairs; p += 8) {
    const int kb0 = p * 32;          // scalar
    if (p + 8 < pairs) {  // scalar branch: prefetch next iteration's K rows
      __builtin_prefetch(kbp + (size_t)((p + 8) * 32 + n16) * DKn, 0, 1);
      __builtin_prefetch(kbp + (size_t)((p + 8) * 32 + 16 + n16) * DKn, 0, 1);
    }
    // two 16x16 score tiles (keys kb0..kb0+15 and kb0+16..kb0+31)
    const _Float16* kr0 = kbp + (size_t)(kb0 + n16) * DKn;
    v16h b0 = ldfrag_h(kr0 + half8,      kr0 + half8 + 16);
    v16h b1 = ldfrag_h(kr0 + 32 + half8, kr0 + 32 + half8 + 16);
    v8f s0 = {};
    s0 = wmma16(aq0, b0, s0);
    s0 = wmma16(aq1, b1, s0);
    const _Float16* kr1 = kbp + (size_t)(kb0 + 16 + n16) * DKn;
    v16h c0 = ldfrag_h(kr1 + half8,      kr1 + half8 + 16);
    v16h c1 = ldfrag_h(kr1 + 32 + half8, kr1 + 32 + half8 + 16);
    v8f s1 = {};
    s1 = wmma16(aq0, c0, s1);
    s1 = wmma16(aq1, c1, s1);

    // scalar (wave-uniform) diagonal test: only ~1 pair per wave needs masks
    if (kb0 + 31 >= qbase) {
#pragma unroll
      for (int r = 0; r < 8; ++r) {
        const int qg = qbase + r + half8;
        float x0 = s0[r];
        if (kb0 + n16 > qg) x0 = -1e9f;
        float x1 = s1[r];
        if (kb0 + 16 + n16 > qg) x1 = -1e9f;
        float mn = fmaxf(mreg[r], rmax16(fmaxf(x0, x1)));
        cacc[r] *= EXP2F(mreg[r] - mn);
        mreg[r] = mn;
        Pbuf[wave][r + half8][n16]      = (_Float16)EXP2F(x0 - mn);
        Pbuf[wave][r + half8][16 + n16] = (_Float16)EXP2F(x1 - mn);
      }
    } else {
#pragma unroll
      for (int r = 0; r < 8; ++r) {
        float x0 = s0[r];
        float x1 = s1[r];
        float mn = fmaxf(mreg[r], rmax16(fmaxf(x0, x1)));
        cacc[r] *= EXP2F(mreg[r] - mn);
        mreg[r] = mn;
        Pbuf[wave][r + half8][n16]      = (_Float16)EXP2F(x0 - mn);
        Pbuf[wave][r + half8][16 + n16] = (_Float16)EXP2F(x1 - mn);
      }
    }
    // re-read P as an A-fragment; ctx += P @ Vh  (col 15 accumulates sum(P))
    const _Float16* prow = &Pbuf[wave][n16][0];
    v16h pa = ldfrag_h(prow + aoff, prow + aoff + 16);
    const _Float16* vrow = vbp + (size_t)n16 * Sn + kb0 + half8;
    v16h vbm = ldfrag_h(vrow, vrow + 16);
    cacc = wmma16(pa, vbm, cacc);
  }

  // publish per-wave maxima
  if ((lane & 15) == 0) {
#pragma unroll
    for (int r = 0; r < 8; ++r) Ms[wave][r + half8] = mreg[r];
  }
  __syncthreads();

  // merge maxima; correct own ctx partial (numerator AND denominator) together
#pragma unroll
  for (int r = 0; r < 8; ++r) {
    const int row = r + half8;
    float m = -1e30f;
#pragma unroll
    for (int w = 0; w < 8; ++w) m = fmaxf(m, Ms[w][row]);
    cacc[r] *= EXP2F(mreg[r] - m);
  }
#pragma unroll
  for (int r = 0; r < 8; ++r) atomicAdd(&Cacc[r + half8][n16], cacc[r]);
  __syncthreads();

  // normalize by the ones-column denominator; write concat buffer [B*S][64]
  {
    const int m = tid >> 4, c = tid & 15;
    const float val = Cacc[m][c] / Cacc[m][15];
    const size_t row = (size_t)b * Sn + qbase + m;
    if (c < DVn) ctx[row * CTXW + h * DVn + c] = (_Float16)val;
    if (h == Hn - 1) ctx[row * CTXW + 48 + c] = (_Float16)0.f;  // zero the pad
  }
}

// ---- K3: output projection  out = ctx @ Wo + bo -----------------------------
__global__ void outproj_kernel(const _Float16* __restrict__ ctx,  // [B*S][64]
                               const _Float16* __restrict__ WoT,  // [512][64]
                               const float* __restrict__ bo,      // [512]
                               float* __restrict__ out) {         // [B*S][512]
  const int lane = threadIdx.x & 31;
  const int wave = __builtin_amdgcn_readfirstlane(threadIdx.x >> 5);  // uniform
  const int NGRP = (En / 16) / 4;     // 8 groups of 4 n-tiles
  const int MT = (Bn * Sn) / 16;      // 512
  int gw = blockIdx.x * (blockDim.x >> 5) + wave;
  if (gw >= MT * NGRP) return;
  const int mt = gw / NGRP, ng = gw % NGRP;
  const int mbase = mt * 16, nbase0 = ng * 64;

  const int n16   = lane & 15;
  const int aoff  = (lane < 16) ? 0 : 8;
  const int boff  = (lane < 16) ? 0 : 16;
  const int half8 = (lane < 16) ? 0 : 8;

  const _Float16* arow = ctx + (size_t)(mbase + n16) * CTXW;

  v8f acc[4];
#pragma unroll
  for (int g = 0; g < 4; ++g) acc[g] = (v8f){};

#pragma unroll
  for (int kb = 0; kb < CTXW; kb += 32) {
    v16h a = ldfrag_h(arow + kb + aoff, arow + kb + aoff + 16);
#pragma unroll
    for (int g = 0; g < 4; ++g) {
      const _Float16* brow = WoT + (size_t)(nbase0 + g * 16 + n16) * CTXW + kb + half8;
      v16h bm = ldfrag_h(brow, brow + 16);
      acc[g] = wmma16(a, bm, acc[g]);
    }
  }
#pragma unroll
  for (int g = 0; g < 4; ++g) {
    const int ncol = nbase0 + g * 16 + n16;
    const float bv = bo[ncol];
#pragma unroll
    for (int r = 0; r < 8; ++r)
      out[(size_t)(mbase + r + half8) * En + ncol] = acc[g][r] + bv;
  }
}

// ---- host-side launcher -----------------------------------------------------
extern "C" void kernel_launch(void* const* d_in, const int* in_sizes, int n_in,
                              void* d_out, int out_size, void* d_ws, size_t ws_size,
                              hipStream_t stream) {
  (void)in_sizes; (void)n_in; (void)out_size;
  const float* Q  = (const float*)d_in[0];
  const float* K  = (const float*)d_in[1];
  const float* V  = (const float*)d_in[2];
  // d_in[3] = mask: causal, handled analytically
  const float* Wq = (const float*)d_in[4];
  const float* bq = (const float*)d_in[5];
  const float* Wk = (const float*)d_in[6];
  const float* bk = (const float*)d_in[7];
  const float* Wv = (const float*)d_in[8];
  const float* bv = (const float*)d_in[9];
  const float* Wo = (const float*)d_in[10];
  const float* bo = (const float*)d_in[11];
  float* out = (float*)d_out;

  char* ws = (char*)d_ws;
  size_t off = 0;
  auto alloc = [&](size_t bytes) {
    char* p = ws + off;
    off += (bytes + 255) & ~(size_t)255;
    return p;
  };
  _Float16* WqT = (_Float16*)alloc((size_t)Hn * DKn * En * 2);
  _Float16* WkT = (_Float16*)alloc((size_t)Hn * DKn * En * 2);
  _Float16* WvT = (_Float16*)alloc((size_t)Hn * DVP * En * 2);
  _Float16* WoT = (_Float16*)alloc((size_t)En * CTXW * 2);
  _Float16* Qh  = (_Float16*)alloc((size_t)Bn * Hn * Sn * DKn * 2);
  _Float16* Kh  = (_Float16*)alloc((size_t)Bn * Hn * Sn * DKn * 2);
  _Float16* VhT = (_Float16*)alloc((size_t)Bn * Hn * DVP * Sn * 2);
  _Float16* ctx = (_Float16*)alloc((size_t)Bn * Sn * CTXW * 2);
  if (ws_size < off) return;  // workspace too small; nothing safe to do

  auto blks = [](int total) { return dim3((unsigned)((total + 255) / 256)); };

  // weights -> f16, transposed/padded for B-fragment loads
  wconv_kernel<<<blks(Hn * DKn * En), 256, 0, stream>>>(Wq, WqT, Hn, En, DKn, En, DKn);
  wconv_kernel<<<blks(Hn * DKn * En), 256, 0, stream>>>(Wk, WkT, Hn, En, DKn, En, DKn);
  wconv_kernel<<<blks(Hn * DVP * En), 256, 0, stream>>>(Wv, WvT, Hn, En, DVn, En, DVP);
  wconv_kernel<<<blks(En * CTXW),     256, 0, stream>>>(Wo, WoT, 1, Hn * DVn, En, CTXW, En);

  // projections: 4096 waves each -> 512 blocks of 8 waves
  proj_kernel<4><<<dim3(512), 256, 0, stream>>>(Q, WqT, bq, Qh, DKn, 0, QSCALE, -1);
  proj_kernel<4><<<dim3(512), 256, 0, stream>>>(K, WkT, bk, Kh, DKn, 0, 1.0f, -1);
  proj_kernel<1><<<dim3(512), 256, 0, stream>>>(V, WvT, bv, VhT, DVn, 1, 1.0f, DVP - 1);

  // causal attention: one block per (b, h, 16-row q tile)
  attn_kernel<<<dim3(Bn * Hn * (Sn / 16)), 256, 0, stream>>>(Qh, Kh, VhT, ctx);

  // output projection: 4096 waves -> 512 blocks
  outproj_kernel<<<dim3(512), 256, 0, stream>>>(ctx, WoT, bo, out);
}
